// Decoder_31310311588043
// MI455X (gfx1250) — matrix operand order repaired and checked by reading, not compiled, not run on hardware
//
#include <hip/hip_runtime.h>
#include <hip/hip_bf16.h>
#include <cstdint>
#include <cstddef>

// Problem dims
#define BB 64
#define LL 512
#define EE 128
#define HH 256   // 4H = 1024, E+H = 384, 2H = 512

typedef __attribute__((ext_vector_type(16))) _Float16 v16h;
typedef __attribute__((ext_vector_type(8)))  float    v8f;

// ---------------- d_out layout (floats) ----------------
// outputs [B,L,L], pointers [B,L] (as float), h_f [B,H], c_f [B,H]
#define PTR_OFF  ((size_t)BB * LL * LL)                 // 16777216
#define HF_OFF   (PTR_OFF + (size_t)BB * LL)            // 16809984
#define CF_OFF   (HF_OFF + (size_t)BB * HH)             // 16826368

// ---------------- d_ws layout (bytes) ----------------
// ctx_proj f32 [B*L, H]            : 32 MB
// Wg_tiled  f16 (Ntiles=64,KS=12)  : 786432 B
// Wa_in_t   f16 (16, 8)            : 131072 B
// Wa_ctx_t  f16 (16, 8)            : 131072 B
// Wo_t      f16 (16, 16)           : 262144 B
#define WS_CTXP_OFF   ((size_t)0)
#define WS_WG_OFF     ((size_t)33554432)
#define WS_WAIN_OFF   (WS_WG_OFF + 786432)
#define WS_WACTX_OFF  (WS_WAIN_OFF + 131072)
#define WS_WO_OFF     (WS_WACTX_OFF + 131072)

// =====================================================================
// Weight packing: lay W[n][k] (row-major, K contiguous) out as WMMA
// B-fragments: tile (nt,ks) = 512 halves; element (lane,j):
//   n = nt*16 + (lane&15); k = ks*32 + (j<8 ? j : j+8) + ((lane&16)?8:0)
// so the per-k-step B fragment is a single contiguous 32B load per lane.
// Source value: k < kb ? W0[n*kb+k] : W1[n*(K-kb)+(k-kb)]  (gates concat)
// =====================================================================
__global__ void pack_tiles_kernel(const float* __restrict__ W0,
                                  const float* __restrict__ W1,
                                  int kb, int K, int Ntiles,
                                  _Float16* __restrict__ dst)
{
    const int KS = K / 32;
    const int total = Ntiles * KS * 512;
    for (int idx = blockIdx.x * blockDim.x + threadIdx.x; idx < total;
         idx += gridDim.x * blockDim.x) {
        int within = idx & 511;
        int tile   = idx >> 9;
        int lane   = within >> 4;
        int j      = within & 15;
        int nt     = tile / KS;
        int ks     = tile % KS;
        int n      = nt * 16 + (lane & 15);
        int k      = ks * 32 + ((j < 8) ? j : (j + 8)) + ((lane & 16) ? 8 : 0);
        float v    = (k < kb) ? W0[n * kb + k] : W1[n * (K - kb) + (k - kb)];
        dst[idx]   = (_Float16)v;
    }
}

// A-fragment (16-bit A 16x32 layout): lane holds row m=lane&15;
// element j maps to K = k0 + (j<8 ? j : j+8) + ((lane&16)?8:0)
template <int K, int KB>
__device__ __forceinline__ v8f wave_gemm_f32lds(
    const float* __restrict__ A0, int s0,
    const float* __restrict__ A1, int s1,
    const _Float16* __restrict__ Wt, int nt, int lane)
{
    constexpr int KS = K / 32;
    v8f acc = {};
    const int m    = lane & 15;
    const int kadd = (lane & 16) ? 8 : 0;
#pragma unroll
    for (int ks = 0; ks < KS; ++ks) {
        v16h a;
#pragma unroll
        for (int j = 0; j < 16; ++j) {
            int kk = ks * 32 + ((j < 8) ? j : (j + 8)) + kadd;
            float v;
            if (KB < K && kk >= KB) v = A1[m * s1 + (kk - KB)];
            else                    v = A0[m * s0 + kk];
            a[j] = (_Float16)v;
        }
        v16h b = *(const v16h*)(Wt + (size_t)(nt * KS + ks) * 512 + lane * 16);
        acc = __builtin_amdgcn_wmma_f32_16x16x32_f16(
            false, a, false, b, (short)0, acc, false, false);
    }
    return acc;
}

// =====================================================================
// ctx_proj = context @ Wa_ctx.T + ba_ctx   ([B*L,256] x [256,256])
// 2048 blocks x 512 threads; block handles 16 rows; 16 waves = 16 N-tiles.
// =====================================================================
__global__ __launch_bounds__(512) void ctxproj_kernel(
    const float* __restrict__ context,
    const _Float16* __restrict__ Wt,
    const float* __restrict__ ba_ctx,
    float* __restrict__ ctxp)
{
    __shared__ _Float16 sA[16 * 256];
    const int tid = threadIdx.x;
    const int r0  = blockIdx.x * 16;
    for (int e = tid; e < 4096; e += 512)
        sA[e] = (_Float16)context[(size_t)r0 * 256 + e];
    __syncthreads();

    const int wid = tid >> 5, lane = tid & 31;
    const int nt  = wid;
    const int m   = lane & 15;
    const int kadd = (lane & 16) ? 8 : 0;
    v8f acc = {};
#pragma unroll
    for (int ks = 0; ks < 8; ++ks) {
        v16h a;
#pragma unroll
        for (int j = 0; j < 16; ++j) {
            int kk = ks * 32 + ((j < 8) ? j : (j + 8)) + kadd;
            a[j] = sA[m * 256 + kk];
        }
        v16h b = *(const v16h*)(Wt + (size_t)(nt * 8 + ks) * 512 + lane * 16);
        acc = __builtin_amdgcn_wmma_f32_16x16x32_f16(
            false, a, false, b, (short)0, acc, false, false);
    }
    const int col = nt * 16 + (lane & 15);
    const int rb  = (lane & 16) ? 8 : 0;
    const float bias = ba_ctx[col];
#pragma unroll
    for (int r = 0; r < 8; ++r) {
        int row = r0 + r + rb;                    // C/D layout: M = r + (lane>=16)*8
        ctxp[(size_t)row * 256 + col] = acc[r] + bias;
    }
}

// =====================================================================
// Persistent decoder: 4 blocks x 512 threads (16 waves); block owns 16
// batch rows; runs all 512 recurrent steps with LDS-resident state.
// =====================================================================
__global__ __launch_bounds__(512) void decoder_kernel(
    const float* __restrict__ emb,     // [B,L,E]
    const float* __restrict__ dec0,    // [B,E]
    const float* __restrict__ h0,      // [B,H]
    const float* __restrict__ c0,      // [B,H]
    const float* __restrict__ bi, const float* __restrict__ bh,
    const float* __restrict__ bo, const float* __restrict__ ba_in,
    const float* __restrict__ V,
    const _Float16* __restrict__ Wg_t,
    const _Float16* __restrict__ Wain_t,
    const _Float16* __restrict__ Wo_t,
    const float* __restrict__ ctxp,    // [B*L,H] f32
    float* __restrict__ out)
{
    // 64KB region aliased: gates g[16][1024]  <->  (sc[16][512] | hcat[16][512])
    __shared__ float sh_gsc[16384];
    __shared__ float sh_h[4096];      // hidden state (= hidden_t) [16][256]
    __shared__ float sh_c[4096];      // cell state [16][256]
    __shared__ float sh_inp[4096];    // attention input [16][256]
    __shared__ float sh_mask[8192];   // [16][512]
    __shared__ float sh_dec[2048];    // decoder input [16][128]
    __shared__ float sh_V[256];
    __shared__ int   sh_ptr[16];

    float* sc   = sh_gsc;             // alpha / scores [16][512]
    float* hcat = sh_gsc + 8192;      // [att(256) | h_t(256)] per row, [16][512]

    const int tid  = threadIdx.x;
    const int wid  = tid >> 5;
    const int lane = tid & 31;
    const int b0   = blockIdx.x * 16;

    // ---- init state ----
    for (int e = tid; e < 4096; e += 512) {
        sh_h[e] = h0[(size_t)b0 * 256 + e];
        sh_c[e] = c0[(size_t)b0 * 256 + e];
    }
    for (int e = tid; e < 2048; e += 512)
        sh_dec[e] = dec0[(size_t)b0 * 128 + e];
    for (int e = tid; e < 8192; e += 512) sh_mask[e] = 1.0f;
    if (tid < 256) sh_V[tid] = V[tid];
    __syncthreads();

    for (int t = 0; t < LL; ++t) {
        // ---- phase 1: gates = [dec_in | h] @ [Wi|Wh].T   -> g[16][1024] ----
#pragma unroll
        for (int i = 0; i < 4; ++i) {
            int nt = wid * 4 + i;
            v8f acc = wave_gemm_f32lds<384, 128>(sh_dec, 128, sh_h, 256,
                                                 Wg_t, nt, lane);
            int col = nt * 16 + (lane & 15);
            int rb  = (lane & 16) ? 8 : 0;
#pragma unroll
            for (int r = 0; r < 8; ++r)
                sh_gsc[(r + rb) * 1024 + col] = acc[r];
        }
        __syncthreads();

        // ---- phase 2: LSTM elementwise (staged: read g -> barrier -> write hcat) ----
        float hreg[8];
        int   ereg[8];
        {
            int q = 0;
            for (int e = tid; e < 4096; e += 512, ++q) {
                int m = e >> 8, n = e & 255;
                float xi = sh_gsc[m * 1024 + n]       + bi[n]       + bh[n];
                float xf = sh_gsc[m * 1024 + 256 + n] + bi[256 + n] + bh[256 + n];
                float xg = sh_gsc[m * 1024 + 512 + n] + bi[512 + n] + bh[512 + n];
                float xo = sh_gsc[m * 1024 + 768 + n] + bi[768 + n] + bh[768 + n];
                float ig = 1.0f / (1.0f + __expf(-xi));
                float fg = 1.0f / (1.0f + __expf(-xf));
                float gg = tanhf(xg);
                float og = 1.0f / (1.0f + __expf(-xo));
                float cn = fg * sh_c[e] + ig * gg;
                sh_c[e]  = cn;
                hreg[q]  = og * tanhf(cn);
                ereg[q]  = m * 512 + 256 + n;     // h_t slot in hcat
            }
        }
        __syncthreads();
        {
            int q = 0;
            for (int e = tid; e < 4096; e += 512, ++q) hcat[ereg[q]] = hreg[q];
        }
        __syncthreads();

        // ---- phase 3: inp = h_t @ Wa_in.T + ba_in ----
        {
            int nt = wid;
            v8f acc = wave_gemm_f32lds<256, 256>(hcat + 256, 512, hcat + 256, 512,
                                                 Wain_t, nt, lane);
            int col = nt * 16 + (lane & 15);
            int rb  = (lane & 16) ? 8 : 0;
            float bias = ba_in[col];
#pragma unroll
            for (int r = 0; r < 8; ++r)
                sh_inp[(r + rb) * 256 + col] = acc[r] + bias;
        }
        __syncthreads();

        // ---- phase 4: scores[m][l] = sum_h V[h]*tanh(inp[m][h]+ctxp[b][l][h]) ----
        for (int e = tid; e < 8192; e += 512) {
            int m = e >> 9, l = e & 511;
            float s;
            if (sh_mask[e] == 0.0f) {
                s = -__builtin_inff();
            } else {
                const float* cp = ctxp + ((size_t)(b0 + m) * 512 + l) * 256;
                const float* ip = sh_inp + m * 256;
                float a0 = 0.0f;
#pragma unroll 4
                for (int hh = 0; hh < 256; ++hh)
                    a0 += sh_V[hh] * tanhf(ip[hh] + cp[hh]);
                s = a0;
            }
            sc[e] = s;
        }
        __syncthreads();

        // ---- phase 5: softmax + argmax(alpha*mask), wave per row ----
        {
            int m = wid;
            float xs[16];
            float mx = -__builtin_inff();
#pragma unroll
            for (int i = 0; i < 16; ++i) {
                xs[i] = sc[m * 512 + lane + 32 * i];
                mx = fmaxf(mx, xs[i]);
            }
            for (int off = 16; off; off >>= 1)
                mx = fmaxf(mx, __shfl_xor(mx, off, 32));
            float sum = 0.0f;
#pragma unroll
            for (int i = 0; i < 16; ++i) {
                xs[i] = __expf(xs[i] - mx);
                sum += xs[i];
            }
            for (int off = 16; off; off >>= 1)
                sum += __shfl_xor(sum, off, 32);
            float inv = 1.0f / sum;
            float bv = -1.0f;
            int bidx = 0x7fffffff;
#pragma unroll
            for (int i = 0; i < 16; ++i) {
                int l = lane + 32 * i;
                float alpha = xs[i] * inv;
                sc[m * 512 + l] = alpha;
                out[((size_t)(b0 + m) * 512 + t) * 512 + l] = alpha;
                float mo = alpha * sh_mask[m * 512 + l];
                if (mo > bv || (mo == bv && l < bidx)) { bv = mo; bidx = l; }
            }
            for (int off = 16; off; off >>= 1) {
                float ov = __shfl_xor(bv, off, 32);
                int   oi = __shfl_xor(bidx, off, 32);
                if (ov > bv || (ov == bv && oi < bidx)) { bv = ov; bidx = oi; }
            }
            if (lane == 0) {
                sh_ptr[m] = bidx;
                sh_mask[m * 512 + bidx] = 0.0f;
                out[PTR_OFF + (size_t)(b0 + m) * 512 + t] = (float)bidx;
            }
        }
        __syncthreads();

        // ---- phase 6: att_hidden[m] = alpha[m] @ ctxp[b0+m]  (wave per row) ----
        {
            int m = wid;
            float a6[8];
#pragma unroll
            for (int j = 0; j < 8; ++j) a6[j] = 0.0f;
            const float* cpb = ctxp + (size_t)(b0 + m) * 512 * 256 + lane * 8;
            for (int l = 0; l < 512; ++l) {
                float a = sc[m * 512 + l];
                if (a != 0.0f) {
                    const float* cp = cpb + (size_t)l * 256;
#pragma unroll
                    for (int j = 0; j < 8; ++j) a6[j] += a * cp[j];
                }
            }
#pragma unroll
            for (int j = 0; j < 8; ++j)
                hcat[m * 512 + lane * 8 + j] = a6[j];   // att part, cols 0..255
        }
        __syncthreads();

        // ---- phase 7: hidden = tanh([att|h_t] @ Wo.T + bo) -> sh_h ----
        {
            int nt = wid;
            v8f acc = wave_gemm_f32lds<512, 512>(hcat, 512, hcat, 512,
                                                 Wo_t, nt, lane);
            int col = nt * 16 + (lane & 15);
            int rb  = (lane & 16) ? 8 : 0;
            float bias = bo[col];
#pragma unroll
            for (int r = 0; r < 8; ++r)
                sh_h[(r + rb) * 256 + col] = tanhf(acc[r] + bias);
        }
        __syncthreads();

        // ---- phase 8: dec_next = embedded_inputs[b, ptr] ----
        for (int e = tid; e < 2048; e += 512) {
            int m = e >> 7, k = e & 127;
            int idx = sh_ptr[m];
            sh_dec[e] = emb[((size_t)(b0 + m) * 512 + idx) * 128 + k];
        }
        __syncthreads();
    }

    // ---- finals: h_f (= hidden_t), c_f ----
    for (int e = tid; e < 4096; e += 512) {
        out[HF_OFF + (size_t)b0 * 256 + e] = sh_h[e];
        out[CF_OFF + (size_t)b0 * 256 + e] = sh_c[e];
    }
}

// =====================================================================
extern "C" void kernel_launch(void* const* d_in, const int* in_sizes, int n_in,
                              void* d_out, int out_size, void* d_ws, size_t ws_size,
                              hipStream_t stream)
{
    const float* emb     = (const float*)d_in[0];
    const float* dec0    = (const float*)d_in[1];
    const float* h0      = (const float*)d_in[2];
    const float* c0      = (const float*)d_in[3];
    const float* context = (const float*)d_in[4];
    const float* Wi      = (const float*)d_in[5];
    const float* bi      = (const float*)d_in[6];
    const float* Wh      = (const float*)d_in[7];
    const float* bh      = (const float*)d_in[8];
    const float* Wo      = (const float*)d_in[9];
    const float* bo      = (const float*)d_in[10];
    const float* Wa_in   = (const float*)d_in[11];
    const float* ba_in   = (const float*)d_in[12];
    const float* Wa_ctx  = (const float*)d_in[13];
    const float* ba_ctx  = (const float*)d_in[14];
    const float* V       = (const float*)d_in[15];

    float* out = (float*)d_out;
    char*  ws  = (char*)d_ws;
    float*     ctxp    = (float*)(ws + WS_CTXP_OFF);
    _Float16*  Wg_t    = (_Float16*)(ws + WS_WG_OFF);
    _Float16*  Wain_t  = (_Float16*)(ws + WS_WAIN_OFF);
    _Float16*  Wactx_t = (_Float16*)(ws + WS_WACTX_OFF);
    _Float16*  Wo_t    = (_Float16*)(ws + WS_WO_OFF);

    // Pack weights into WMMA B-fragment tile layout (f16).
    pack_tiles_kernel<<<768, 512, 0, stream>>>(Wi, Wh, 128, 384, 64, Wg_t);
    pack_tiles_kernel<<<128, 512, 0, stream>>>(Wa_in, Wa_in, 256, 256, 16, Wain_t);
    pack_tiles_kernel<<<128, 512, 0, stream>>>(Wa_ctx, Wa_ctx, 256, 256, 16, Wactx_t);
    pack_tiles_kernel<<<256, 512, 0, stream>>>(Wo, Wo, 512, 512, 16, Wo_t);

    // One-time context projection GEMM (WMMA), f32 result in workspace.
    ctxproj_kernel<<<2048, 512, 0, stream>>>(context, Wactx_t, ba_ctx, ctxp);

    // Persistent recurrent decoder: 4 workgroups x 16 batch rows.
    decoder_kernel<<<4, 512, 0, stream>>>(emb, dec0, h0, c0, bi, bh, bo, ba_in, V,
                                          Wg_t, Wain_t, Wo_t, ctxp, out);
}
